// FastSingleHeadAttn_18837726561081
// MI455X (gfx1250) — matrix operand order repaired and checked by reading, not compiled, not run on hardware
//
#include <hip/hip_runtime.h>
#include <stdint.h>

typedef __attribute__((ext_vector_type(16))) __bf16 v16bf;
typedef __attribute__((ext_vector_type(8)))  float  v8f;

#define TILE_M 128
#define TILE_N 128
#define TILE_K 32
#define A_LD   36    // bf16 elems per As row: 72 B (8B aligned), dword stride 18 -> conflict-free
#define B_LD   132   // bf16 elems per Bs row: 264 B (8B aligned), dword stride 66 -> conflict-free

union BF2  { __bf16 h[2]; uint32_t u; };
union Frag { v16bf v; uint32_t u[8]; };

// MODE 0: A fp32 [M,K], B fp32 [N,K] (NT), +bias, C bf16    (QKV projection)
// MODE 1: A bf16 [M,K], B bf16 [N,K] (NT), *scale, C fp32   (scores Q*K^T)
// MODE 2: A bf16 [M,K], B bf16 [K,N] (NN),          C fp32  (P*V)
template<int MODE>
__global__ __launch_bounds__(256)
void gemm_wmma_kernel(const void* __restrict__ Ag, const void* __restrict__ Bg,
                      const float* __restrict__ bias, void* __restrict__ Cg,
                      int K, int lda, int ldb, int ldc,
                      long long sA, long long sB, long long sC,
                      float scale, int nbn)
{
    __shared__ __align__(16) __bf16 As[2][TILE_M * A_LD];   // 2 x 9216 B
    __shared__ __align__(16) __bf16 Bs[2][TILE_K * B_LD];   // 2 x 8448 B

    const int tid   = threadIdx.x;
    const int lane  = tid & 31;
    const int wave  = tid >> 5;
    const int waveM = wave >> 1;            // 0..3 : 32 rows each
    const int waveN = wave & 1;             // 0..1 : 64 cols each
    const int bm    = (int)blockIdx.x / nbn;
    const int bn    = (int)blockIdx.x % nbn;
    const int m0    = bm * TILE_M;
    const int n0    = bn * TILE_N;
    const long long batch = blockIdx.z;

    v8f acc[2][4] = {};

    float4 aF[4], bF[4];   // MODE 0 staging regs
    uint2  aU[4], bU[4];   // MODE 1/2 staging regs

    // ---- issue all global loads for one K-slab into registers (max MLP) ----
    auto loadGlobal = [&](int k0) {
        if constexpr (MODE == 0) {
            const float* A = (const float*)Ag;
            const float* B = (const float*)Bg;
            #pragma unroll
            for (int i = 0; i < 4; ++i) {
                int idx = tid + 256 * i;            // 128 rows x 8 float4-cols
                int r = idx >> 3, c = idx & 7;
                aF[i] = *(const float4*)(A + (long long)(m0 + r) * lda + k0 + 4 * c);
                bF[i] = *(const float4*)(B + (long long)(n0 + r) * ldb + k0 + 4 * c);
            }
        } else if constexpr (MODE == 1) {
            const __bf16* A = (const __bf16*)Ag + batch * sA;
            const __bf16* B = (const __bf16*)Bg + batch * sB;
            #pragma unroll
            for (int i = 0; i < 4; ++i) {
                int idx = tid + 256 * i;            // 128 rows x 8 uint2-cols (4 bf16 each)
                int r = idx >> 3, c = idx & 7;
                aU[i] = *(const uint2*)(A + (long long)(m0 + r) * lda + k0 + 4 * c);
                bU[i] = *(const uint2*)(B + (long long)(n0 + r) * ldb + k0 + 4 * c);
            }
        } else {
            const __bf16* A = (const __bf16*)Ag + batch * sA;
            const __bf16* B = (const __bf16*)Bg + batch * sB;
            #pragma unroll
            for (int i = 0; i < 4; ++i) {
                int idx = tid + 256 * i;
                int r = idx >> 3, c = idx & 7;      // A: 128 rows x 8 cols
                aU[i] = *(const uint2*)(A + (long long)(m0 + r) * lda + k0 + 4 * c);
                int kk = idx >> 5, cc = idx & 31;   // B: 32 k-rows x 32 uint2-cols
                bU[i] = *(const uint2*)(B + (long long)(k0 + kk) * ldb + n0 + 4 * cc);
            }
        }
    };

    // ---- convert (if fp32) + store staged registers into LDS buffer ----
    auto storeLds = [&](int buf) {
        __bf16* Asb = As[buf];
        __bf16* Bsb = Bs[buf];
        if constexpr (MODE == 0) {
            #pragma unroll
            for (int i = 0; i < 4; ++i) {
                int idx = tid + 256 * i;
                int r = idx >> 3, c = idx & 7;
                BF2 t0, t1;
                t0.h[0] = (__bf16)aF[i].x; t0.h[1] = (__bf16)aF[i].y;
                t1.h[0] = (__bf16)aF[i].z; t1.h[1] = (__bf16)aF[i].w;
                *(uint2*)(Asb + r * A_LD + 4 * c) = make_uint2(t0.u, t1.u);
                Bsb[(4 * c + 0) * B_LD + r] = (__bf16)bF[i].x;  // transpose [N,K] -> Bs[k][n]
                Bsb[(4 * c + 1) * B_LD + r] = (__bf16)bF[i].y;
                Bsb[(4 * c + 2) * B_LD + r] = (__bf16)bF[i].z;
                Bsb[(4 * c + 3) * B_LD + r] = (__bf16)bF[i].w;
            }
        } else if constexpr (MODE == 1) {
            #pragma unroll
            for (int i = 0; i < 4; ++i) {
                int idx = tid + 256 * i;
                int r = idx >> 3, c = idx & 7;
                *(uint2*)(Asb + r * A_LD + 4 * c) = aU[i];
                BF2 lo, hi; lo.u = bU[i].x; hi.u = bU[i].y;
                Bsb[(4 * c + 0) * B_LD + r] = lo.h[0];
                Bsb[(4 * c + 1) * B_LD + r] = lo.h[1];
                Bsb[(4 * c + 2) * B_LD + r] = hi.h[0];
                Bsb[(4 * c + 3) * B_LD + r] = hi.h[1];
            }
        } else {
            #pragma unroll
            for (int i = 0; i < 4; ++i) {
                int idx = tid + 256 * i;
                int r = idx >> 3, c = idx & 7;
                *(uint2*)(Asb + r * A_LD + 4 * c) = aU[i];
                int kk = idx >> 5, cc = idx & 31;
                *(uint2*)(Bsb + kk * B_LD + 4 * cc) = bU[i];   // direct copy [K,N]
            }
        }
    };

    const int nk = K / TILE_K;
    loadGlobal(0);

    for (int t = 0; t < nk; ++t) {
        const int buf = t & 1;
        storeLds(buf);
        __syncthreads();
        if (t + 1 < nk) loadGlobal((t + 1) * TILE_K);   // overlap with WMMA below

        const uint32_t* As32 = (const uint32_t*)As[buf];
        const uint32_t* Bs32 = (const uint32_t*)Bs[buf];

        Frag af[2]; Frag bfr[4];
        const int mlow = lane & 15, half = lane >> 4;
        #pragma unroll
        for (int mt = 0; mt < 2; ++mt) {
            int mrow = waveM * 32 + mt * 16 + mlow;
            #pragma unroll
            for (int p = 0; p < 8; ++p) {
                int kk = (p >> 2) * 16 + half * 8 + (p & 3) * 2;   // ISA 16-bit A 16x32 layout
                af[mt].u[p] = As32[mrow * (A_LD / 2) + (kk >> 1)];
            }
        }
        #pragma unroll
        for (int nt = 0; nt < 4; ++nt) {
            int ncol = waveN * 64 + nt * 16 + mlow;                // lane indexes N, elems index K (same order as A)
            #pragma unroll
            for (int p = 0; p < 8; ++p) {
                int kk = (p >> 2) * 16 + half * 8 + (p & 3) * 2;
                BF2 tb;
                tb.h[0] = Bs[buf][kk * B_LD + ncol];
                tb.h[1] = Bs[buf][(kk + 1) * B_LD + ncol];
                bfr[nt].u[p] = tb.u;
            }
        }
        #pragma unroll
        for (int mt = 0; mt < 2; ++mt)
            #pragma unroll
            for (int nt = 0; nt < 4; ++nt)
                acc[mt][nt] = __builtin_amdgcn_wmma_f32_16x16x32_bf16(
                    false, af[mt].v, false, bfr[nt].v,
                    (short)0, acc[mt][nt], false, false);
        // no trailing barrier: double-buffered
    }

    // ---------------- epilogue: D layout m = v + 8*half, n = lane&15 ----------------
    const int nlow = lane & 15, halfe = lane >> 4;
    #pragma unroll
    for (int mt = 0; mt < 2; ++mt)
        #pragma unroll
        for (int nt = 0; nt < 4; ++nt) {
            const int col = n0 + waveN * 64 + nt * 16 + nlow;
            #pragma unroll
            for (int v = 0; v < 8; ++v) {
                const int row = m0 + waveM * 32 + mt * 16 + v + 8 * halfe;
                float val = acc[mt][nt][v];
                if constexpr (MODE == 0) {
                    val += bias[col];
                    ((__bf16*)Cg)[(long long)row * ldc + col] = (__bf16)val;
                } else if constexpr (MODE == 1) {
                    ((float*)Cg + batch * sC)[(long long)row * ldc + col] = val * scale;
                } else {
                    ((float*)Cg + batch * sC)[(long long)row * ldc + col] = val;
                }
            }
        }
}

// Row softmax over 2048 fp32 scores; writes bf16 probs in-place (pitch 4096 bf16).
__global__ __launch_bounds__(256)
void softmax_kernel(float* __restrict__ S)
{
    const long long row = blockIdx.x;                 // 0 .. 8*2048-1
    float* srow = S + row * 2048;
    const float4* s4 = (const float4*)srow;
    uint2* p2 = (uint2*)srow;                         // bf16 out, first 4 KB of this row
    const int tid = threadIdx.x;

    float4 v0 = s4[tid];
    float4 v1 = s4[tid + 256];
    float mx = fmaxf(fmaxf(fmaxf(v0.x, v0.y), fmaxf(v0.z, v0.w)),
                     fmaxf(fmaxf(v1.x, v1.y), fmaxf(v1.z, v1.w)));

    __shared__ float red[256];
    red[tid] = mx;
    __syncthreads();
    for (int s = 128; s > 0; s >>= 1) {
        if (tid < s) red[tid] = fmaxf(red[tid], red[tid + s]);
        __syncthreads();
    }
    mx = red[0];
    __syncthreads();

    v0.x = __expf(v0.x - mx); v0.y = __expf(v0.y - mx);
    v0.z = __expf(v0.z - mx); v0.w = __expf(v0.w - mx);
    v1.x = __expf(v1.x - mx); v1.y = __expf(v1.y - mx);
    v1.z = __expf(v1.z - mx); v1.w = __expf(v1.w - mx);
    red[tid] = (v0.x + v0.y + v0.z + v0.w) + (v1.x + v1.y + v1.z + v1.w);
    __syncthreads();
    for (int s = 128; s > 0; s >>= 1) {
        if (tid < s) red[tid] += red[tid + s];
        __syncthreads();
    }
    const float inv = 1.0f / red[0];                  // all reads done: in-place write safe

    BF2 a, b, c, d;
    a.h[0] = (__bf16)(v0.x * inv); a.h[1] = (__bf16)(v0.y * inv);
    b.h[0] = (__bf16)(v0.z * inv); b.h[1] = (__bf16)(v0.w * inv);
    c.h[0] = (__bf16)(v1.x * inv); c.h[1] = (__bf16)(v1.y * inv);
    d.h[0] = (__bf16)(v1.z * inv); d.h[1] = (__bf16)(v1.w * inv);
    p2[tid]       = make_uint2(a.u, b.u);             // bf16 elems 4*tid   .. 4*tid+3
    p2[tid + 256] = make_uint2(c.u, d.u);             // bf16 elems 1024+4*tid ..
}

extern "C" void kernel_launch(void* const* d_in, const int* in_sizes, int n_in,
                              void* d_out, int out_size, void* d_ws, size_t ws_size,
                              hipStream_t stream)
{
    (void)in_sizes; (void)n_in; (void)out_size; (void)ws_size;
    const float* x = (const float*)d_in[0];     // [8,2048,768]
    const float* W = (const float*)d_in[1];     // [2304,768]
    const float* b = (const float*)d_in[2];     // [2304]
    float* out = (float*)d_out;                 // [8,2048,768]

    char* ws = (char*)d_ws;
    __bf16* qkv = (__bf16*)ws;                                   // [16384,2304] bf16 (75.5 MB)
    float*  S   = (float*)(ws + (long long)16384 * 2304 * 2);    // [8,2048,2048] f32 (134 MB), P aliased

    const float scale = 0.036084391824351614f;  // 768^-0.5

    // 1) qkv = x @ W^T + b
    gemm_wmma_kernel<0><<<dim3(128 * 18, 1, 1), 256, 0, stream>>>(
        x, W, b, qkv, 768, 768, 768, 2304, 0, 0, 0, 1.0f, 18);

    // 2) S[b] = scale * Q[b] @ K[b]^T
    gemm_wmma_kernel<1><<<dim3(16 * 16, 1, 8), 256, 0, stream>>>(
        qkv, qkv + 768, nullptr, S, 768, 2304, 2304, 2048,
        (long long)2048 * 2304, (long long)2048 * 2304, (long long)2048 * 2048, scale, 16);

    // 3) row softmax -> bf16 P in place (row pitch 4096 bf16)
    softmax_kernel<<<dim3(8 * 2048), 256, 0, stream>>>(S);

    // 4) out[b] = P[b] @ V[b]
    gemm_wmma_kernel<2><<<dim3(16 * 6, 1, 8), 256, 0, stream>>>(
        (const void*)S, qkv + 1536, nullptr, out, 2048, 4096, 2304, 768,
        (long long)2048 * 4096, (long long)2048 * 2304, (long long)2048 * 768, 1.0f, 6);
}
